// RouterLinear_62740882260717
// MI455X (gfx1250) — matrix operand order repaired and checked by reading, not compiled, not run on hardware
//
#include <hip/hip_runtime.h>

// Problem constants (from reference): B, IN, OUT, K, TOPK
#define B_DIM   1024
#define IN_DIM  4096
#define OUT_DIM 4096
#define K_DIM   64
#define TOPK    64

// GEMM tiling
#define M_BLK   128              // samples per block
#define N_BLK   128              // outputs per block
#define K_BLK   64               // K elements staged per TDM chunk
#define NCHUNK  (IN_DIM / K_BLK) // 64
#define LDS_STRIDE 72            // 64 halves + 8 pad halves (TDM pad: 4 DW / 32 DW)

typedef _Float16 v4h  __attribute__((ext_vector_type(4)));
typedef _Float16 v8h  __attribute__((ext_vector_type(8)));
typedef _Float16 v16h __attribute__((ext_vector_type(16)));
typedef float    v8f  __attribute__((ext_vector_type(8)));
typedef unsigned int u32x4 __attribute__((ext_vector_type(4)));
typedef int      i32x4 __attribute__((ext_vector_type(4)));
typedef int      i32x8 __attribute__((ext_vector_type(8)));

#if defined(__has_builtin)
#  if __has_builtin(__builtin_amdgcn_tensor_load_to_lds)
#    define HAVE_TDM 1
#  endif
#endif
#ifndef HAVE_TDM
#  define HAVE_TDM 0
#endif

// ---------------------------------------------------------------------------
// 1) zero the dense scatter buffer Xs (f32)
// ---------------------------------------------------------------------------
__global__ void rl_zero_f32(float* __restrict__ p, int n) {
    int i = blockIdx.x * blockDim.x + threadIdx.x;
    if (i < n) p[i] = 0.0f;
}

// ---------------------------------------------------------------------------
// 2) scatter-add: Xs[b, prev_idx[b,k]] += x[b,k]
// ---------------------------------------------------------------------------
__global__ void rl_scatter(const float* __restrict__ x,
                           const int* __restrict__ idx,
                           float* __restrict__ Xs) {
    int g = blockIdx.x * blockDim.x + threadIdx.x;   // 0 .. B*K-1
    int b = g >> 6;                                  // /K_DIM
    atomicAdd(&Xs[(size_t)b * IN_DIM + idx[g]], x[g]);
}

// ---------------------------------------------------------------------------
// 3) f32 -> f16 conversion (vectorized x4)
// ---------------------------------------------------------------------------
__global__ void rl_cvt_f16(const float* __restrict__ src,
                           _Float16* __restrict__ dst, int n4) {
    int i = blockIdx.x * blockDim.x + threadIdx.x;
    if (i < n4) {
        float4 v = ((const float4*)src)[i];
        v4h h = { (_Float16)v.x, (_Float16)v.y, (_Float16)v.z, (_Float16)v.w };
        ((v4h*)dst)[i] = h;
    }
}

// ---------------------------------------------------------------------------
// TDM: DMA one [tile_dim1=128 rows] x [tile_dim0=64 halves] tile into LDS,
// padding 4 DWORDs after every 32 DWORDs -> LDS row stride = 72 halves.
// Descriptor per CDNA5 ISA Ch.8 (D# groups 0/1; groups 2/3 unused for 2D).
// ---------------------------------------------------------------------------
#if HAVE_TDM
__device__ __forceinline__ void tdm_load_tile(const _Float16* gsrc,
                                              unsigned lds_byte_off) {
    unsigned long long ga = (unsigned long long)(size_t)gsrc;
    u32x4 g0;
    g0.x = 1u;                                             // count=1, user desc
    g0.y = lds_byte_off;                                   // lds_addr
    g0.z = (unsigned)ga;                                   // global_addr[31:0]
    g0.w = (unsigned)((ga >> 32) & 0x01FFFFFFull)          // global_addr[56:32]
         | (2u << 30);                                     // type = 2 (image)
    i32x8 g1;
    g1[0] = (1 << 16)        // data_size = 1 -> 2 bytes
          | (1 << 20)        // pad_enable
          | (4 << 22)        // pad_interval code 4 -> every 32 DWORDs
          | (3 << 25);       // pad_amount  code 3 -> 4 DWORDs (8 halves)
    g1[1] = (int)((IN_DIM & 0xFFFF) << 16);                // tensor_dim0 lo
    g1[2] = (int)(((unsigned)IN_DIM >> 16)                 // tensor_dim0 hi
          | ((OUT_DIM & 0xFFFF) << 16));                   // tensor_dim1 lo
    g1[3] = (int)(((unsigned)OUT_DIM >> 16)                // tensor_dim1 hi
          | (K_BLK << 16));                                // tile_dim0 = 64
    g1[4] = N_BLK;                                         // tile_dim1=128, tile_dim2=0
    g1[5] = IN_DIM;                                        // tensor_dim0_stride lo32
    g1[6] = 0;                                             // stride hi | dim1_stride lo
    g1[7] = 0;
    i32x4 z4 = {0, 0, 0, 0};
#if __clang_major__ >= 23
    i32x8 z8 = {0, 0, 0, 0, 0, 0, 0, 0};
    __builtin_amdgcn_tensor_load_to_lds(g0, g1, z4, z4, z8, 0);
#else
    __builtin_amdgcn_tensor_load_to_lds(g0, g1, z4, z4, 0);
#endif
}
#endif

// ---------------------------------------------------------------------------
// one K_BLK=64 chunk of wave-level WMMA work: A direct from global,
// B fragments from the LDS-staged (padded) tile.
// ---------------------------------------------------------------------------
__device__ __forceinline__ void rl_compute_chunk(
    const _Float16* __restrict__ Xsh, const _Float16* __restrict__ sBc,
    int aRowBase, int i0, int wn, int l15, int hl, int kbase, v8f (&acc)[2][4])
{
#pragma unroll
    for (int ks = 0; ks < 2; ++ks) {
        const int kk = i0 + ks * 32;
        // A: lane<16 holds K {0..7,16..23}, lane>=16 holds {8..15,24..31}
        v16h afrag[2];
#pragma unroll
        for (int m = 0; m < 2; ++m) {
            const _Float16* ap =
                Xsh + (size_t)(aRowBase + m * 16 + l15) * IN_DIM + kk + kbase;
            v8h lo = *(const v8h*)(ap);
            v8h hi = *(const v8h*)(ap + 16);
            afrag[m] = __builtin_shufflevector(lo, hi,
                0,1,2,3,4,5,6,7,8,9,10,11,12,13,14,15);
            __builtin_prefetch(ap + 64, 0, 3);   // next chunk, WGP scope
        }
        // B from LDS: lane = output column, K {0..7,16..23} / {8..15,24..31}
        v16h bfrag[4];
#pragma unroll
        for (int n = 0; n < 4; ++n) {
            const _Float16* bp =
                sBc + (wn * 64 + n * 16 + l15) * LDS_STRIDE + ks * 32 + hl * 8;
            v8h lo = *(const v8h*)(bp);
            v8h hi = *(const v8h*)(bp + 16);
            bfrag[n] = __builtin_shufflevector(lo, hi,
                0,1,2,3,4,5,6,7,8,9,10,11,12,13,14,15);
        }
#pragma unroll
        for (int m = 0; m < 2; ++m)
#pragma unroll
            for (int n = 0; n < 4; ++n)
                acc[m][n] = __builtin_amdgcn_wmma_f32_16x16x32_f16(
                    false, afrag[m], false, bfrag[n],
                    (short)0, acc[m][n], false, false);
    }
}

// ---------------------------------------------------------------------------
// 4) WMMA GEMM: result[b,o] = sum_i Xsh[b,i]*Wh[o,i] + bias[o]
//    Block 256 thr = 8 waves; wave tile 32x64; block tile 128x128.
//    B tile double-buffered in LDS via TDM (tensor_load_to_lds), pipelined
//    with s_wait_tensorcnt; A streamed directly from global (L2-hot, 2x reuse).
// ---------------------------------------------------------------------------
__launch_bounds__(256)
__global__ void rl_gemm_wmma(const _Float16* __restrict__ Xsh,  // B   x IN
                             const _Float16* __restrict__ Wh,   // OUT x IN
                             const float* __restrict__ bias,    // OUT
                             float* __restrict__ out)           // B x OUT
{
    __shared__ _Float16 sB[2][N_BLK * LDS_STRIDE];   // 2 x 18 KB

    const int lane = threadIdx.x & 31;
    const int wave = threadIdx.x >> 5;     // 0..7
    const int wm   = wave >> 1;            // 0..3 -> M sub-tile
    const int wn   = wave & 1;             // 0..1 -> N sub-tile
    const int hl   = lane >> 4;
    const int l15  = lane & 15;
    const int kbase = hl * 8;

    const int mBase = blockIdx.y * M_BLK;
    const int nBase = blockIdx.x * N_BLK;
    const int aRowBase = mBase + wm * 32;

    v8f acc[2][4] = {};

#if HAVE_TDM
    const bool issuer = (wave == 0);
    if (issuer) {
        tdm_load_tile(Wh + (size_t)nBase * IN_DIM,
                      (unsigned)(unsigned long long)(const void*)&sB[0][0]);
    }
    for (int kc = 0; kc < NCHUNK; ++kc) {
        const int cur = kc & 1;
        if (issuer) {
            if (kc + 1 < NCHUNK) {
                tdm_load_tile(Wh + (size_t)nBase * IN_DIM + (kc + 1) * K_BLK,
                    (unsigned)(unsigned long long)(const void*)&sB[cur ^ 1][0]);
                __builtin_amdgcn_s_wait_tensorcnt(1);   // chunk kc landed
            } else {
                __builtin_amdgcn_s_wait_tensorcnt(0);
            }
        }
        __syncthreads();                                 // publish buf[cur]
        rl_compute_chunk(Xsh, sB[cur], aRowBase, kc * K_BLK,
                         wn, l15, hl, kbase, acc);
        __syncthreads();                                 // done reading buf[cur]
    }
#else
    // Fallback: cooperative single-buffer staging (no TDM builtin available)
    for (int kc = 0; kc < NCHUNK; ++kc) {
        __syncthreads();
        for (int c = threadIdx.x; c < (N_BLK * K_BLK) / 8; c += 256) {
            const int row = c >> 3;
            const int col = (c & 7) * 8;
            *(v8h*)(&sB[0][0] + row * LDS_STRIDE + col) =
                *(const v8h*)(Wh + (size_t)(nBase + row) * IN_DIM
                              + kc * K_BLK + col);
        }
        __syncthreads();
        rl_compute_chunk(Xsh, sB[0], aRowBase, kc * K_BLK,
                         wn, l15, hl, kbase, acc);
    }
#endif

    // C/D layout: element r -> row (r + 8*hl), col l15
#pragma unroll
    for (int m = 0; m < 2; ++m) {
#pragma unroll
        for (int n = 0; n < 4; ++n) {
            const int o  = nBase + wn * 64 + n * 16 + l15;
            const float bv = bias[o];
#pragma unroll
            for (int r = 0; r < 8; ++r) {
                const int brow = aRowBase + m * 16 + r + hl * 8;
                out[(size_t)brow * OUT_DIM + o] = acc[m][n][r] + bv;
            }
        }
    }
}

// ---------------------------------------------------------------------------
// 5) top-64 per sample: iterative LDS argmax (descending, ties -> lower index)
// ---------------------------------------------------------------------------
__launch_bounds__(256)
__global__ void rl_topk(const float* __restrict__ result,
                        float* __restrict__ vals,
                        int* __restrict__ sel) {
    __shared__ float svals[OUT_DIM];   // 16 KB
    __shared__ float rv[256];
    __shared__ int   ri[256];

    const int b   = blockIdx.x;
    const int tid = threadIdx.x;
    const float NEG = -3.402823466e38f;

    for (int j = tid; j < OUT_DIM; j += 256)
        svals[j] = result[(size_t)b * OUT_DIM + j];
    __syncthreads();

    for (int t = 0; t < TOPK; ++t) {
        float best = NEG;
        int   bi   = OUT_DIM;
        for (int j = tid; j < OUT_DIM; j += 256) {
            float v = svals[j];
            if (v > best) { best = v; bi = j; }
        }
        rv[tid] = best; ri[tid] = bi;
        __syncthreads();
        for (int s = 128; s > 0; s >>= 1) {
            if (tid < s) {
                float ov = rv[tid + s]; int oi = ri[tid + s];
                if (ov > rv[tid] || (ov == rv[tid] && oi < ri[tid])) {
                    rv[tid] = ov; ri[tid] = oi;
                }
            }
            __syncthreads();
        }
        if (tid == 0) {
            vals[(size_t)b * TOPK + t] = rv[0];
            sel [(size_t)b * TOPK + t] = ri[0];
            svals[ri[0]] = NEG;
        }
        __syncthreads();
    }
}

// ---------------------------------------------------------------------------
// launch
// ---------------------------------------------------------------------------
extern "C" void kernel_launch(void* const* d_in, const int* in_sizes, int n_in,
                              void* d_out, int out_size, void* d_ws, size_t ws_size,
                              hipStream_t stream) {
    const float* x        = (const float*)d_in[0];   // B*K
    const float* weight   = (const float*)d_in[1];   // OUT*IN
    const float* bias     = (const float*)d_in[2];   // OUT
    const int*   prev_idx = (const int*)d_in[3];     // B*K
    // d_in[4] = top_k scalar (compile-time TOPK here)

    char* ws = (char*)d_ws;
    // Workspace layout (56 MB total):
    //   [0,16MB)   Xs f32 (B x IN)  -- reused as result f32 (B x OUT) after cvt
    //   [16,24MB)  Xsh f16 (B x IN)
    //   [24,56MB)  Wh  f16 (OUT x IN)
    float*    Xs     = (float*)ws;
    _Float16* Xsh    = (_Float16*)(ws + (size_t)16 * 1024 * 1024);
    _Float16* Wh     = (_Float16*)(ws + (size_t)24 * 1024 * 1024);
    float*    result = Xs;   // region reuse: Xs is dead once Xsh is built

    const int nXs = B_DIM * IN_DIM;            // 4,194,304
    const int nW  = OUT_DIM * IN_DIM;          // 16,777,216
    const int nBK = B_DIM * K_DIM;             // 65,536

    rl_zero_f32<<<nXs / 256, 256, 0, stream>>>(Xs, nXs);
    rl_scatter<<<nBK / 256, 256, 0, stream>>>(x, prev_idx, Xs);
    rl_cvt_f16<<<(nXs / 4) / 256, 256, 0, stream>>>(Xs, Xsh, nXs / 4);
    rl_cvt_f16<<<(nW / 4) / 256, 256, 0, stream>>>(weight, Wh, nW / 4);

    dim3 ggrid(OUT_DIM / N_BLK, B_DIM / M_BLK);    // (32, 8)
    rl_gemm_wmma<<<ggrid, 256, 0, stream>>>(Xsh, Wh, bias, result);

    float* vals = (float*)d_out;
    int*   sel  = (int*)d_out + (size_t)B_DIM * TOPK;
    rl_topk<<<B_DIM, 256, 0, stream>>>(result, vals, sel);
}
